// HybridAttention_60739427500724
// MI455X (gfx1250) — matrix operand order repaired and checked by reading, not compiled, not run on hardware
//
#include <hip/hip_runtime.h>

// ---------------------------------------------------------------------------
// Types
// ---------------------------------------------------------------------------
typedef __bf16 bf16;
typedef __attribute__((ext_vector_type(16))) __bf16 v16bf;
typedef __attribute__((ext_vector_type(8)))  __bf16 v8bf;
typedef __attribute__((ext_vector_type(8)))  float  v8f;
typedef unsigned int v4u __attribute__((ext_vector_type(4)));
typedef int          v4i __attribute__((ext_vector_type(4)));
typedef int          v8i __attribute__((ext_vector_type(8)));

#define T_SEQ   2048
#define D_MODEL 1024
#define N_HEADS 16
#define D_HEAD  64
#define K_AUG   96     // 64 head dims + 6 plucker dims padded to 3*32

// ---------------------------------------------------------------------------
// f32 -> bf16 elementwise convert
// ---------------------------------------------------------------------------
__global__ void cvt_bf16_kernel(const float* __restrict__ src,
                                bf16* __restrict__ dst, int n) {
    for (int i = blockIdx.x * blockDim.x + threadIdx.x; i < n;
         i += gridDim.x * blockDim.x)
        dst[i] = (bf16)src[i];
}

// W[K,N] f32 (row-major) -> Wt[N,K] bf16 (row-major)
__global__ void transpose_cvt_kernel(const float* __restrict__ src,
                                     bf16* __restrict__ dst, int K, int N) {
    int total = K * N;
    for (int i = blockIdx.x * blockDim.x + threadIdx.x; i < total;
         i += gridDim.x * blockDim.x) {
        int k = i / N, n = i % N;
        dst[(size_t)n * K + k] = (bf16)src[i];
    }
}

// ---------------------------------------------------------------------------
// Helpers
// ---------------------------------------------------------------------------
// A 16x32 frag in contiguous k order (B must match): lane holds row M=l16; halfword e -> K=16*half+e.
// Works on global or LDS-backed pointers.
__device__ inline v16bf load_a_frag(const bf16* row, int half) {
    v8bf lo = *(const v8bf*)(row + 16 * half);
    v8bf hi = *(const v8bf*)(row + 16 * half + 8);
    v16bf a;
    #pragma unroll
    for (int e = 0; e < 8; ++e) { a[e] = lo[e]; a[8 + e] = hi[e]; }
    return a;
}

// Issue a TDM DMA of a rows(n) x 32(k) bf16 tile from a row-major matrix with
// leading dimension K (tile top-left at gbase) into LDS at byte offset
// lds_addr.  D# layout per CDNA5 ISA ch.8: group0 = {flags/count, lds_addr,
// global_addr, type=2}; group1 = {data_size, tensor dims, tile dims, stride}.
__device__ inline void tdm_load_panel(const bf16* gbase, int K, int rows,
                                      unsigned lds_addr) {
    unsigned long long ga = (unsigned long long)(uintptr_t)gbase;
    v4u g0;
    g0[0] = 1u;                                          // count=1, user mode
    g0[1] = lds_addr;                                    // LDS dest (bytes)
    g0[2] = (unsigned)ga;                                // global addr [31:0]
    g0[3] = (unsigned)((ga >> 32) & 0x1FFFFFFu) | (2u << 30); // [56:32]|type=2
    v8i g1;
    g1[0] = (1 << 16);       // workgroup_mask=0, data_size=1 (2 bytes)
    g1[1] = (32 << 16);      // tensor_dim0[15:0] = 32 (k extent)
    g1[2] = (rows << 16);    // tensor_dim0 hi = 0; tensor_dim1[15:0] = rows
    g1[3] = (32 << 16);      // tensor_dim1 hi = 0; tile_dim0 = 32
    g1[4] = rows;            // tile_dim1 = rows; tile_dim2 = 0
    g1[5] = K;               // tensor_dim0_stride low32 (elements)
    g1[6] = 0;               // stride hi / dim1_stride lo
    g1[7] = 0;
    v4i gz = {0, 0, 0, 0};
#if __clang_major__ >= 23
    v8i gz8 = {0, 0, 0, 0, 0, 0, 0, 0};
    __builtin_amdgcn_tensor_load_to_lds(g0, g1, gz, gz, gz8, 0);
#else
    __builtin_amdgcn_tensor_load_to_lds(g0, g1, gz, gz, 0);
#endif
}

// ---------------------------------------------------------------------------
// bf16 WMMA GEMM, fully TDM-staged:
//   C[M,N] f32 = A[M,K] @ Bt[N,K]^T (+ bias[n])
// 128 threads = 4 waves; block tile = 128(M) x 64(N); wave tile = 32 x 64.
// Per k-step both the 128x32 A panel and the 64x32 B panel are DMA'd into
// double-buffered LDS by wave 0; the hot loop is ds_load + WMMA only, with
// the next pair of DMAs in flight (s_wait_tensorcnt <= 2 retires the pair
// belonging to the current step).
// ---------------------------------------------------------------------------
__global__ __launch_bounds__(128) void gemm_bf16_tdm_kernel(
    const bf16* __restrict__ A,    // [M,K]
    const bf16* __restrict__ Bt,   // [N,K]
    const float* __restrict__ bias,// [N] or nullptr
    float* __restrict__ C,         // [M,N]
    int M, int N, int K) {
    __shared__ bf16 apanel[2][128 * 32];  // 16 KB double buffer (A)
    __shared__ bf16 bpanel[2][64 * 32];   //  8 KB double buffer (B)

    const int wave = threadIdx.x >> 5;
    const int lane = threadIdx.x & 31;
    const int half = lane >> 4;
    const int l16  = lane & 15;
    const int m0 = blockIdx.x * 128 + wave * 32;
    const int n0 = blockIdx.y * 64;

    const v8f vz = {0.f, 0.f, 0.f, 0.f, 0.f, 0.f, 0.f, 0.f};
    v8f acc[2][4] = {{vz, vz, vz, vz}, {vz, vz, vz, vz}};

    const bf16* apan = A  + (size_t)(blockIdx.x * 128) * K;
    const bf16* bpan = Bt + (size_t)n0 * K;

    const unsigned ldsA0 = (unsigned)(uintptr_t)&apanel[0][0];
    const unsigned ldsA1 = (unsigned)(uintptr_t)&apanel[1][0];
    const unsigned ldsB0 = (unsigned)(uintptr_t)&bpanel[0][0];
    const unsigned ldsB1 = (unsigned)(uintptr_t)&bpanel[1][0];

    const int numIt = K >> 5;   // K/32 k-steps

    if (wave == 0) {
        tdm_load_panel(apan, K, 128, ldsA0);
        tdm_load_panel(bpan, K, 64,  ldsB0);
    }

    for (int it = 0; it < numIt; ++it) {
        const int cur = it & 1;
        if (wave == 0) {
            if (it + 1 < numIt) {
                // issue next A/B pair, then wait for the current pair
                tdm_load_panel(apan + (it + 1) * 32, K, 128, cur ? ldsA0 : ldsA1);
                tdm_load_panel(bpan + (it + 1) * 32, K, 64,  cur ? ldsB0 : ldsB1);
                __builtin_amdgcn_s_wait_tensorcnt((short)2);
            } else {
                __builtin_amdgcn_s_wait_tensorcnt((short)0);
            }
        }
        __syncthreads();                 // panels[cur] visible to all waves

        const bf16* abuf = &apanel[cur][0];
        const bf16* bbuf = &bpanel[cur][0];
        v16bf a0 = load_a_frag(abuf + (wave * 32 + l16) * 32, half);
        v16bf a1 = load_a_frag(abuf + (wave * 32 + 16 + l16) * 32, half);

        #pragma unroll
        for (int f = 0; f < 4; ++f) {
            // B frag: lane holds col n = f*16+l16; K = 16*half + e
            v16bf b = *(const v16bf*)(bbuf + (f * 16 + l16) * 32 + half * 16);
            acc[0][f] = __builtin_amdgcn_wmma_f32_16x16x32_bf16(
                false, a0, false, b, (short)0, acc[0][f], false, false);
            acc[1][f] = __builtin_amdgcn_wmma_f32_16x16x32_bf16(
                false, a1, false, b, (short)0, acc[1][f], false, false);
        }
        __syncthreads();                 // done reading panels[cur]
    }

    #pragma unroll
    for (int s = 0; s < 2; ++s) {
        #pragma unroll
        for (int f = 0; f < 4; ++f) {
            int n = n0 + f * 16 + l16;
            float bv = bias ? bias[n] : 0.0f;
            #pragma unroll
            for (int r = 0; r < 8; ++r) {
                int m = m0 + s * 16 + r + 8 * half;
                C[(size_t)m * N + n] = acc[s][f][r] + bv;
            }
        }
    }
}

// ---------------------------------------------------------------------------
// Pack: build augmented q/k (bf16, K=96) and transposed V from f32 GEMM outs.
// One thread per (h, t).
// ---------------------------------------------------------------------------
__global__ void pack_kernel(const float* __restrict__ qkvf,  // [T, 3*D]
                            const float* __restrict__ pf,    // [T, 128]
                            const float* __restrict__ pscale,// [H]
                            bf16* __restrict__ q_aug,        // [H][T][96]
                            bf16* __restrict__ k_aug,        // [H][T][96]
                            bf16* __restrict__ vT) {         // [H][64][T]
    int idx = blockIdx.x * blockDim.x + threadIdx.x;
    if (idx >= N_HEADS * T_SEQ) return;
    int h = idx & (N_HEADS - 1);
    int t = idx >> 4;

    const float rs = 0.125f;  // dh^-0.5 folded into q
    bf16* qa = q_aug + ((size_t)h * T_SEQ + t) * K_AUG;
    bf16* ka = k_aug + ((size_t)h * T_SEQ + t) * K_AUG;
    const float* src = qkvf + (size_t)t * (3 * D_MODEL) + h * D_HEAD;
    #pragma unroll 4
    for (int d = 0; d < D_HEAD; ++d) {
        qa[d] = (bf16)(src[d] * rs);
        ka[d] = (bf16)(src[D_MODEL + d]);
        vT[((size_t)h * D_HEAD + d) * T_SEQ + t] = (bf16)(src[2 * D_MODEL + d]);
    }

    float p1[4], p2[4];
    #pragma unroll
    for (int c = 0; c < 4; ++c) {
        p1[c] = pf[t * 128 + h * 4 + c];
        p2[c] = pf[t * 128 + 64 + h * 4 + c];
    }
    float l[6];
    l[0] = p1[0] * p2[1] - p1[1] * p2[0];
    l[1] = p1[0] * p2[2] - p1[2] * p2[0];
    l[2] = p1[0] * p2[3] - p1[3] * p2[0];
    l[3] = p1[1] * p2[2] - p1[2] * p2[1];
    l[4] = p1[1] * p2[3] - p1[3] * p2[1];
    l[5] = p1[2] * p2[3] - p1[3] * p2[2];
    float nn = sqrtf(l[0]*l[0] + l[1]*l[1] + l[2]*l[2] +
                     l[3]*l[3] + l[4]*l[4] + l[5]*l[5]);
    float inv = 1.0f / fmaxf(nn, 1e-12f);
    #pragma unroll
    for (int c = 0; c < 6; ++c) l[c] *= inv;
    float Jl[6] = { l[5], -l[4], l[3], l[2], -l[1], l[0] };
    float sc = pscale[h];
    #pragma unroll
    for (int c = 0; c < 6; ++c) {
        qa[D_HEAD + c] = (bf16)(sc * l[c]);   // fold plucker_scale into q side
        ka[D_HEAD + c] = (bf16)(Jl[c]);
    }
    #pragma unroll
    for (int c = 6; c < 32; ++c) {
        qa[D_HEAD + c] = (bf16)0.0f;
        ka[D_HEAD + c] = (bf16)0.0f;
    }
}

// ---------------------------------------------------------------------------
// Causal flash attention, 1 wave per 16 query rows per head.
// Computes S^T = K_aug x Q_aug^T so the probability C-fragment converts
// lane-locally into the P A-fragment for P@V (no cross-lane transpose).
// ---------------------------------------------------------------------------
__global__ __launch_bounds__(32) void flash_kernel(
    const bf16* __restrict__ q_aug,  // [H][T][96]
    const bf16* __restrict__ k_aug,  // [H][T][96]
    const bf16* __restrict__ vT,     // [H][64][T]
    bf16* __restrict__ ctxb) {       // [T][D]
    const int iblk = blockIdx.x;     // 0..127 (16 query rows each)
    const int h    = blockIdx.y;     // 0..15
    const int lane = threadIdx.x;
    const int half = lane >> 4;
    const int l16  = lane & 15;
    const int ig   = iblk * 16 + l16;  // global query row owned by this lane

    const v8f vz = {0.f, 0.f, 0.f, 0.f, 0.f, 0.f, 0.f, 0.f};

    // Q as B-fragments (K=c, N=i): lane holds col i=l16, c = kc*32+16*half+e
    const bf16* qrow = q_aug + ((size_t)h * T_SEQ + iblk * 16 + l16) * K_AUG;
    v16bf qb[3];
    #pragma unroll
    for (int kc = 0; kc < 3; ++kc)
        qb[kc] = *(const v16bf*)(qrow + kc * 32 + 16 * half);

    float m_i = -1e30f, l_i = 0.0f;
    v8f o[4] = {vz, vz, vz, vz};

    const int num_jb = (iblk >> 1) + 1;  // 32-wide key blocks (causal)
    for (int b = 0; b < num_jb; ++b) {
        const int jb = b * 32;

        // --- S^T tiles: T0 rows j=jb..jb+15, T1 rows jb+16..jb+31 ----------
        const bf16* krow0 = k_aug + ((size_t)h * T_SEQ + jb + l16) * K_AUG;
        const bf16* krow1 = krow0 + 16 * K_AUG;
        v8f S0 = vz, S1 = vz;
        #pragma unroll
        for (int kc = 0; kc < 3; ++kc) {
            v16bf a0 = load_a_frag(krow0 + kc * 32, half);
            v16bf a1 = load_a_frag(krow1 + kc * 32, half);
            S0 = __builtin_amdgcn_wmma_f32_16x16x32_bf16(
                false, a0, false, qb[kc], (short)0, S0, false, false);
            S1 = __builtin_amdgcn_wmma_f32_16x16x32_bf16(
                false, a1, false, qb[kc], (short)0, S1, false, false);
        }

        // --- online softmax (lane owns query row i = ig) -------------------
        float s0[8], s1[8];
        float lm = -1e30f;
        #pragma unroll
        for (int r = 0; r < 8; ++r) {
            int j0 = jb + 8 * half + r;        // C frag row r+8*half = key j
            int j1 = j0 + 16;
            s0[r] = (j0 <= ig) ? S0[r] : -1e30f;  // causal mask
            s1[r] = (j1 <= ig) ? S1[r] : -1e30f;
            lm = fmaxf(lm, fmaxf(s0[r], s1[r]));
        }
        lm = fmaxf(lm, __shfl_xor(lm, 16, 32));  // merge j-halves
        float mnew  = fmaxf(m_i, lm);
        float alpha = __expf(m_i - mnew);
        float rsum  = 0.0f;
        v16bf pa;  // P A-fragment, built lane-locally from S^T C-fragments
        #pragma unroll
        for (int r = 0; r < 8; ++r) {
            float e0 = __expf(s0[r] - mnew);
            float e1 = __expf(s1[r] - mnew);
            rsum += e0 + e1;
            pa[r]     = (bf16)e0;   // K = jb + 8*half + r
            pa[8 + r] = (bf16)e1;   // K = jb + 16 + 8*half + r
        }
        rsum += __shfl_xor(rsum, 16, 32);
        l_i = l_i * alpha + rsum;
        m_i = mnew;

        // rescale O accumulators (rows live in VGPR index -> broadcast alpha)
        #pragma unroll
        for (int r = 0; r < 8; ++r) {
            float a = __shfl(alpha, r + 8 * half, 32);
            o[0][r] *= a; o[1][r] *= a; o[2][r] *= a; o[3][r] *= a;
        }

        // --- O += P @ V ----------------------------------------------------
        #pragma unroll
        for (int f = 0; f < 4; ++f) {
            const bf16* vrow =
                vT + ((size_t)h * D_HEAD + f * 16 + l16) * T_SEQ + jb + 8 * half;
            v8bf vlo = *(const v8bf*)vrow;
            v8bf vhi = *(const v8bf*)(vrow + 16);
            v16bf vb = __builtin_shufflevector(vlo, vhi, 0, 1, 2, 3, 4, 5, 6, 7, 8, 9, 10, 11, 12, 13, 14, 15);
            o[f] = __builtin_amdgcn_wmma_f32_16x16x32_bf16(
                false, pa, false, vb, (short)0, o[f], false, false);
        }
    }

    // normalize and store context (bf16 for the output-projection WMMA GEMM)
    float linv = 1.0f / l_i;
    #pragma unroll
    for (int r = 0; r < 8; ++r) {
        float li = __shfl(linv, r + 8 * half, 32);
        int m = iblk * 16 + r + 8 * half;
        bf16* crow = ctxb + (size_t)m * D_MODEL + h * D_HEAD;
        #pragma unroll
        for (int f = 0; f < 4; ++f)
            crow[f * 16 + l16] = (bf16)(o[f][r] * li);
    }
}

// ---------------------------------------------------------------------------
// Launch
// ---------------------------------------------------------------------------
extern "C" void kernel_launch(void* const* d_in, const int* in_sizes, int n_in,
                              void* d_out, int out_size, void* d_ws, size_t ws_size,
                              hipStream_t stream) {
    (void)in_sizes; (void)n_in; (void)out_size; (void)ws_size;
    const float* x     = (const float*)d_in[0];  // [1,2048,1024]
    const float* Wqkv  = (const float*)d_in[1];  // [1024,3072]
    const float* bqkv  = (const float*)d_in[2];  // [3072]
    const float* W1p   = (const float*)d_in[3];  // [1024,64]
    const float* W2p   = (const float*)d_in[4];  // [1024,64]
    const float* pscal = (const float*)d_in[5];  // [16]
    const float* Wout  = (const float*)d_in[6];  // [1024,1024]
    const float* bout  = (const float*)d_in[7];  // [1024]
    float* out = (float*)d_out;                  // [1,2048,1024]

    char* ws = (char*)d_ws;
    bf16*  xb    = (bf16*)(ws + 0);          //  4 MB  x bf16 [2048,1024]
    bf16*  wqkvt = (bf16*)(ws + 4194304);    //  6 MB  Wqkv^T bf16 [3072,1024]
    bf16*  wpt   = (bf16*)(ws + 10485760);   // 256 KB [W1p|W2p]^T bf16 [128,1024]
    bf16*  woutt = (bf16*)(ws + 10747904);   //  2 MB  Wout^T bf16 [1024,1024]
    float* qkvf  = (float*)(ws + 12845056);  // 24 MB  qkv f32 [2048,3072]
    float* pf    = (float*)(ws + 38010880);  //  1 MB  p f32 [2048,128]
    bf16*  qaug  = (bf16*)(ws + 39059456);   //  6 MB  [16,2048,96]
    bf16*  kaug  = (bf16*)(ws + 45350912);   //  6 MB  [16,2048,96]
    bf16*  vT    = (bf16*)(ws + 51642368);   //  4 MB  [16,64,2048]
    bf16*  ctxb  = (bf16*)(ws + 55836672);   //  4 MB  [2048,1024]

    cvt_bf16_kernel<<<dim3(8192), 256, 0, stream>>>(x, xb, T_SEQ * D_MODEL);
    transpose_cvt_kernel<<<dim3(12288), 256, 0, stream>>>(Wqkv, wqkvt, D_MODEL, 3 * D_MODEL);
    transpose_cvt_kernel<<<dim3(256), 256, 0, stream>>>(W1p, wpt, D_MODEL, 64);
    transpose_cvt_kernel<<<dim3(256), 256, 0, stream>>>(W2p, wpt + 64 * D_MODEL, D_MODEL, 64);
    transpose_cvt_kernel<<<dim3(4096), 256, 0, stream>>>(Wout, woutt, D_MODEL, D_MODEL);

    // qkv = x @ Wqkv + bqkv   (block tile 128M x 64N)
    gemm_bf16_tdm_kernel<<<dim3(16, 48), 128, 0, stream>>>(
        xb, wqkvt, bqkv, qkvf, T_SEQ, 3 * D_MODEL, D_MODEL);
    // p = x @ [W1p|W2p]
    gemm_bf16_tdm_kernel<<<dim3(16, 2), 128, 0, stream>>>(
        xb, wpt, nullptr, pf, T_SEQ, 128, D_MODEL);

    pack_kernel<<<dim3(128), 256, 0, stream>>>(qkvf, pf, pscal, qaug, kaug, vT);

    flash_kernel<<<dim3(T_SEQ / 16, N_HEADS), 32, 0, stream>>>(qaug, kaug, vT, ctxb);

    // out = ctx @ Wout + bout
    gemm_bf16_tdm_kernel<<<dim3(16, 16), 128, 0, stream>>>(
        ctxb, woutt, bout, out, T_SEQ, D_MODEL, D_MODEL);
}
